// SimpleMambaBlock_81527069213345
// MI455X (gfx1250) — matrix operand order repaired and checked by reading, not compiled, not run on hardware
//
#include <hip/hip_runtime.h>
#include <math.h>

#define D_DIM   1024
#define B_SZ    8
#define T_LEN   2048
#define M_TOT   (B_SZ * T_LEN)      // 16384 rows
#define LN_EPS  1e-5f
#define SCAN_NBLK 64                // 64 workgroups * 16 cols = 1024 cols

typedef __bf16 v16bf __attribute__((ext_vector_type(16)));
typedef float  v8f   __attribute__((ext_vector_type(8)));

union Frag {
    v16bf v;
    uint4 q[2];
};
static_assert(sizeof(Frag) == 32, "frag must be 32B");

// fp32 -> bf16 round-to-nearest-even
__device__ __forceinline__ unsigned short f2bf(float f) {
    unsigned int u = __float_as_uint(f);
    unsigned int r = u + 0x7FFFu + ((u >> 16) & 1u);
    return (unsigned short)(r >> 16);
}

// LDS byte offset of a __shared__ pointer (generic LDS address low 32 bits are
// the LDS offset per the flat-aperture rules).
__device__ __forceinline__ unsigned lds_off(const void* p) {
    return (unsigned)(unsigned long long)p;
}

// Async global -> LDS copy, 16B per lane, tracked on ASYNCcnt.
__device__ __forceinline__ void async_to_lds_b128(unsigned lds, const void* gaddr) {
    asm volatile("global_load_async_to_lds_b128 %0, %1, off"
                 :: "v"(lds), "v"(gaddr) : "memory");
}
// Device-scope variant (bypass potentially-stale WGP$ for cross-WG data).
__device__ __forceinline__ void async_to_lds_b128_dev(unsigned lds, const void* gaddr) {
    asm volatile("global_load_async_to_lds_b128 %0, %1, off scope:SCOPE_DEV"
                 :: "v"(lds), "v"(gaddr) : "memory");
}
__device__ __forceinline__ void wait_asynccnt0() {
    asm volatile("s_wait_asynccnt 0x0" ::: "memory");
}

// ---------------------------------------------------------------------------
// fp32 -> bf16 conversion pre-pass (row-major copy; the B-fragment this code loads from a
// row-major weight is a contiguous per-lane K-run, so no swizzle is needed).
// ---------------------------------------------------------------------------
__global__ void pack_bf16_kernel(const float* __restrict__ src,
                                 unsigned short* __restrict__ dst, int n) {
    int i = blockIdx.x * blockDim.x + threadIdx.x;
    int stride = gridDim.x * blockDim.x;
    for (; i < n; i += stride) dst[i] = f2bf(src[i]);
}

__global__ void init_bar_kernel(unsigned int* bar) {
    bar[0] = 0u;   // arrive count
    bar[1] = 0u;   // generation
}

// ---------------------------------------------------------------------------
// Fused projection GEMM: for each 16x16 tile of [M_TOT x D], compute
//   Bt = x*Wb^T + bb ; Ct = x*Wc^T + bc ; Gt = sigmoid(x*Wg^T + bg)
//   drive = Bt * Gt
// One wave per tile; A-fragment (x) reused across three WMMAs per K-chunk.
// Outputs stored in scan-friendly [T][8][D] layout.
// ---------------------------------------------------------------------------
__global__ void proj_kernel(const unsigned short* __restrict__ xbf,  // [M_TOT][D]
                            const unsigned short* __restrict__ wb,   // [D][D]
                            const unsigned short* __restrict__ wc,
                            const unsigned short* __restrict__ wg,
                            const float* __restrict__ bb,
                            const float* __restrict__ bc,
                            const float* __restrict__ bg,
                            float* __restrict__ driveBuf,            // [T][8][D]
                            float* __restrict__ cBuf)                // [T][8][D]
{
    const int lane  = threadIdx.x & 31;
    const int wave  = threadIdx.x >> 5;
    const int tile  = blockIdx.x * 8 + wave;   // 65536 tiles total
    const int mtile = tile >> 6;               // / (D/16)
    const int ntile = tile & 63;
    const int m0 = mtile * 16, n0 = ntile * 16;
    const int lmod = lane & 15, lhalf = lane >> 4;

    v8f accB = {}, accC = {}, accG = {};

    const unsigned short* xrow = xbf + (size_t)(m0 + lmod) * D_DIM;
    const int nrow = n0 + lmod;
    const unsigned short* wbp = wb + (size_t)nrow * D_DIM + lhalf * 16;
    const unsigned short* wcp = wc + (size_t)nrow * D_DIM + lhalf * 16;
    const unsigned short* wgp = wg + (size_t)nrow * D_DIM + lhalf * 16;

    for (int k0 = 0; k0 < D_DIM; k0 += 32) {
        Frag a, fb, fc, fg;
        // A-fragment: contiguous 16-elem K-run per lane (matches B)
        a.q[0] = *(const uint4*)(xrow + k0 + lhalf * 16);
        a.q[1] = *(const uint4*)(xrow + k0 + lhalf * 16 + 8);
        // B-fragments: contiguous 16-elem K-run per lane (row-major weight)
        const uint4* pb = (const uint4*)(wbp + k0);
        fb.q[0] = pb[0]; fb.q[1] = pb[1];
        const uint4* pc = (const uint4*)(wcp + k0);
        fc.q[0] = pc[0]; fc.q[1] = pc[1];
        const uint4* pg = (const uint4*)(wgp + k0);
        fg.q[0] = pg[0]; fg.q[1] = pg[1];

        accB = __builtin_amdgcn_wmma_f32_16x16x32_bf16(false, a.v, false, fb.v,
                                                       (short)0, accB, false, false);
        accC = __builtin_amdgcn_wmma_f32_16x16x32_bf16(false, a.v, false, fc.v,
                                                       (short)0, accC, false, false);
        accG = __builtin_amdgcn_wmma_f32_16x16x32_bf16(false, a.v, false, fg.v,
                                                       (short)0, accG, false, false);
    }

    const int n = n0 + lmod;
    const float biasB = bb[n], biasC = bc[n], biasG = bg[n];
#pragma unroll
    for (int r = 0; r < 8; r++) {
        // C/D layout: lane holds rows m = r + 8*(lane/16), col n = lane%16
        int m = m0 + r + 8 * lhalf;
        int t = m & (T_LEN - 1);
        int b = m >> 11;                 // / T_LEN
        float vB = accB[r] + biasB;
        float vG = accG[r] + biasG;
        float vC = accC[r] + biasC;
        float drive = vB * (1.0f / (1.0f + __expf(-vG)));
        size_t idx = (size_t)(t * B_SZ + b) * D_DIM + n;
        driveBuf[idx] = drive;
        cBuf[idx]     = vC;
    }
}

// ---------------------------------------------------------------------------
// Sequential scan: h_t = tanh(h_{t-1} @ A^T + drive_t), y_t = c_t * h_t.
// All 8 batch rows ride in one 16-row WMMA A-operand (rows 8..15 zero pad).
// 64 workgroups each own 16 output columns. The 32 KB weight slice for those
// columns is loaded ONCE into LDS via async-to-LDS and stays resident for all
// 2048 steps; per step both WMMA operands come from ds_load_b128. The full h
// state is broadcast through a double-buffered global bf16 buffer; each step
// ends with a device-scope split barrier (plus s_cluster_barrier: HW barrier
// under cluster dispatch, architectural NOP otherwise) followed by an async
// global->LDS reload of h at device scope (ASYNCcnt / s_wait_asynccnt).
// ---------------------------------------------------------------------------
__global__ void scan_kernel(const unsigned short* __restrict__ Abf,  // [D][D] bf16 (row n, col k)
                            const float* __restrict__ driveBuf,      // [T][8][D]
                            const float* __restrict__ cBuf,          // [T][8][D]
                            unsigned short* __restrict__ hglob,      // [2][16][D] bf16 ping-pong
                            float* __restrict__ y,                   // [B][T][D] (= d_out)
                            unsigned int* __restrict__ bar)          // [0]=cnt [1]=gen
{
    __shared__ unsigned short sA[16 * D_DIM];   // 32 KB: weight slice (resident)
    __shared__ unsigned short sH[16 * D_DIM];   // 32 KB: current h, bf16

    const int lane  = threadIdx.x & 31;
    const int wave  = threadIdx.x >> 5;
    const int lmod  = lane & 15, lhalf = lane >> 4;
    const bool compute = (wave == 0);           // one 16x16 N-tile per block
    const int n0g = blockIdx.x * 16;            // global col base of tile

    // ---- one-time: async-fill weight slice sA[n_local][k] = Abf[n0g+n_local][k]
    {
        const unsigned short* src = Abf + (size_t)n0g * D_DIM;  // 32 KB contiguous
        for (int i = threadIdx.x; i < 16 * D_DIM / 8; i += blockDim.x)  // 2048 x 16B
            async_to_lds_b128(lds_off(&sA[i * 8]), src + i * 8);
    }
    // ---- h0 = 0
    {
        uint4 z = {0u, 0u, 0u, 0u};
        uint4* p = (uint4*)sH;
        for (int i = threadIdx.x; i < 16 * D_DIM / 8; i += blockDim.x) p[i] = z;
    }
    wait_asynccnt0();
    __syncthreads();

    for (int t = 0; t < T_LEN; t++) {
        unsigned short* hout = hglob + ((t + 1) & 1) * 16 * D_DIM;

        if (compute) {
            const int n = n0g + lmod;
            v8f acc = {};
#pragma unroll
            for (int r = 0; r < 8; r++) {
                int m = r + 8 * lhalf;
                acc[r] = (m < 8) ? driveBuf[(size_t)(t * B_SZ + m) * D_DIM + n] : 0.0f;
            }
            const unsigned short* wp = sA + lmod * D_DIM + lhalf * 16;  // B-frag (LDS)
            const unsigned short* hp = sH + lmod * D_DIM + lhalf * 16;  // A-frag (LDS)
            for (int k0 = 0; k0 < D_DIM; k0 += 32) {
                Frag a, b;
                a.q[0] = *(const uint4*)(hp + k0);         // ds_load_b128
                a.q[1] = *(const uint4*)(hp + k0 + 8);     // ds_load_b128
                const uint4* pw = (const uint4*)(wp + k0);
                b.q[0] = pw[0]; b.q[1] = pw[1];            // ds_load_b128 x2
                acc = __builtin_amdgcn_wmma_f32_16x16x32_bf16(false, a.v, false, b.v,
                                                              (short)0, acc, false, false);
            }
#pragma unroll
            for (int r = 0; r < 8; r++) {
                int m = r + 8 * lhalf;
                float h = tanhf(acc[r]);                   // pad rows stay exactly 0
                hout[m * D_DIM + n] = f2bf(h);
                if (m < 8) {
                    float c = cBuf[(size_t)(t * B_SZ + m) * D_DIM + n];
                    y[(size_t)(m * T_LEN + t) * D_DIM + n] = c * h;
                }
            }
        }

        // prefetch next step's drive slice (global_prefetch_b8)
        if (t + 1 < T_LEN) {
            const float* pf = driveBuf +
                (size_t)((t + 1) * B_SZ + (threadIdx.x & 7)) * D_DIM + blockIdx.x * 16;
            __builtin_prefetch(pf, 0, 1);
        }

        // ------- device-wide split barrier between steps -------
        __threadfence();
        __syncthreads();
        if (threadIdx.x == 0) {
            unsigned int g = __hip_atomic_load(&bar[1], __ATOMIC_RELAXED,
                                               __HIP_MEMORY_SCOPE_AGENT);
            unsigned int prev = __hip_atomic_fetch_add(&bar[0], 1u, __ATOMIC_ACQ_REL,
                                                       __HIP_MEMORY_SCOPE_AGENT);
            if (prev == SCAN_NBLK - 1) {
                __hip_atomic_store(&bar[0], 0u, __ATOMIC_RELAXED,
                                   __HIP_MEMORY_SCOPE_AGENT);
                __hip_atomic_fetch_add(&bar[1], 1u, __ATOMIC_RELEASE,
                                       __HIP_MEMORY_SCOPE_AGENT);
            } else {
                while (__hip_atomic_load(&bar[1], __ATOMIC_ACQUIRE,
                                         __HIP_MEMORY_SCOPE_AGENT) == g) {
                    __builtin_amdgcn_s_sleep(1);
                }
            }
            // Real HW cluster barrier when dispatched as a cluster;
            // architecturally S_NOP when ClusterID==0.
            __builtin_amdgcn_s_cluster_barrier();
        }
        __syncthreads();

        // reload full h state: async global->LDS at device scope (fresh data)
        for (int i = threadIdx.x; i < 16 * D_DIM / 8; i += blockDim.x)   // 2048 x 16B
            async_to_lds_b128_dev(lds_off(&sH[i * 8]), hout + i * 8);
        wait_asynccnt0();
        __syncthreads();
    }
}

// ---------------------------------------------------------------------------
// In-place LayerNorm over the last dim of y (=d_out): one block per row.
// ---------------------------------------------------------------------------
__global__ void ln_kernel(float* __restrict__ y,
                          const float* __restrict__ gamma,
                          const float* __restrict__ beta)
{
    __shared__ float ssum[256], ssq[256];
    const int row = blockIdx.x;
    const int c = threadIdx.x * 4;
    float4 v = *(const float4*)(y + (size_t)row * D_DIM + c);
    ssum[threadIdx.x] = v.x + v.y + v.z + v.w;
    ssq[threadIdx.x]  = v.x * v.x + v.y * v.y + v.z * v.z + v.w * v.w;
    __syncthreads();
    for (int off = 128; off > 0; off >>= 1) {
        if (threadIdx.x < off) {
            ssum[threadIdx.x] += ssum[threadIdx.x + off];
            ssq[threadIdx.x]  += ssq[threadIdx.x + off];
        }
        __syncthreads();
    }
    float mu  = ssum[0] * (1.0f / D_DIM);
    float var = ssq[0] * (1.0f / D_DIM) - mu * mu;
    float rs  = rsqrtf(var + LN_EPS);
    float4 g  = *(const float4*)(gamma + c);
    float4 be = *(const float4*)(beta + c);
    float4 o;
    o.x = (v.x - mu) * rs * g.x + be.x;
    o.y = (v.y - mu) * rs * g.y + be.y;
    o.z = (v.z - mu) * rs * g.z + be.z;
    o.w = (v.w - mu) * rs * g.w + be.w;
    *(float4*)(y + (size_t)row * D_DIM + c) = o;
}

// ---------------------------------------------------------------------------
extern "C" void kernel_launch(void* const* d_in, const int* in_sizes, int n_in,
                              void* d_out, int out_size, void* d_ws, size_t ws_size,
                              hipStream_t stream)
{
    (void)in_sizes; (void)n_in; (void)out_size; (void)ws_size;

    const float* x     = (const float*)d_in[0];
    const float* A     = (const float*)d_in[1];
    const float* Wb    = (const float*)d_in[2];
    const float* bb    = (const float*)d_in[3];
    const float* Wc    = (const float*)d_in[4];
    const float* bc    = (const float*)d_in[5];
    const float* Wg    = (const float*)d_in[6];
    const float* bg    = (const float*)d_in[7];
    const float* gamma = (const float*)d_in[8];
    const float* beta  = (const float*)d_in[9];
    float* out = (float*)d_out;

    char* ws = (char*)d_ws;
    size_t off = 0;
    auto alloc = [&](size_t bytes) -> char* {
        char* p = ws + off;
        off = (off + bytes + 255) & ~(size_t)255;
        return p;
    };
    unsigned int*   bar      = (unsigned int*)  alloc(256);
    unsigned short* xbf      = (unsigned short*)alloc((size_t)M_TOT * D_DIM * 2);
    unsigned short* wbbf     = (unsigned short*)alloc((size_t)D_DIM * D_DIM * 2);
    unsigned short* wcbf     = (unsigned short*)alloc((size_t)D_DIM * D_DIM * 2);
    unsigned short* wgbf     = (unsigned short*)alloc((size_t)D_DIM * D_DIM * 2);
    unsigned short* abf      = (unsigned short*)alloc((size_t)D_DIM * D_DIM * 2);
    float*          driveBuf = (float*)         alloc((size_t)T_LEN * B_SZ * D_DIM * 4);
    float*          cBuf     = (float*)         alloc((size_t)T_LEN * B_SZ * D_DIM * 4);
    unsigned short* hglob    = (unsigned short*)alloc((size_t)2 * 16 * D_DIM * 2);

    init_bar_kernel<<<1, 1, 0, stream>>>(bar);

    pack_bf16_kernel<<<4096, 256, 0, stream>>>(x,  xbf,  M_TOT * D_DIM);
    pack_bf16_kernel<<<1024, 256, 0, stream>>>(Wb, wbbf, D_DIM * D_DIM);
    pack_bf16_kernel<<<1024, 256, 0, stream>>>(Wc, wcbf, D_DIM * D_DIM);
    pack_bf16_kernel<<<1024, 256, 0, stream>>>(Wg, wgbf, D_DIM * D_DIM);
    pack_bf16_kernel<<<1024, 256, 0, stream>>>(A,  abf,  D_DIM * D_DIM);

    // 65536 tiles (1024 M-tiles x 64 N-tiles), 8 waves/block
    proj_kernel<<<8192, 256, 0, stream>>>(xbf, wbbf, wcbf, wgbf, bb, bc, bg,
                                          driveBuf, cBuf);

    scan_kernel<<<SCAN_NBLK, 128, 0, stream>>>(abf, driveBuf, cBuf, hglob, out, bar);

    ln_kernel<<<M_TOT, 256, 0, stream>>>(out, gamma, beta);
}
